// TransA_7653631721898
// MI455X (gfx1250) — matrix operand order repaired and checked by reading, not compiled, not run on hardware
//
#include <hip/hip_runtime.h>
#include <hip/hip_bf16.h>

// Problem constants (match reference)
#define ENT_SIZE 1000000
#define REL_SIZE 1000
#define EMB_DIM  64
#define BATCH    8192

#define JTILES   (BATCH / 16)          // 512 j-tiles of 16 rows
#define NTASK    (REL_SIZE * JTILES)   // 512,000 wave-tasks
#define WAVES_PER_BLOCK 8              // 256 threads = 8 wave32

typedef __bf16 v16bf __attribute__((ext_vector_type(16)));
typedef float  v8f   __attribute__((ext_vector_type(8)));

// ---------------------------------------------------------------------------
// Kernel 1: e[j][d] = | ent[head_j][d] + relemb[rel_j][d] - ent[tail_j][d] |
// ---------------------------------------------------------------------------
__global__ __launch_bounds__(256) void err_kernel(
    const int* __restrict__ trip,       // (BATCH, 3) int32
    const float* __restrict__ ent,      // (ENT_SIZE, 64)
    const float* __restrict__ rele,     // (REL_SIZE, 64)
    float* __restrict__ e32)            // (BATCH, 64) out
{
    const int idx = blockIdx.x * 256 + threadIdx.x;   // 0 .. BATCH*64-1
    const int j = idx >> 6;
    const int d = idx & 63;
    const long h = trip[j * 3 + 0];
    const long r = trip[j * 3 + 1];
    const long t = trip[j * 3 + 2];
    const float v = ent[h * EMB_DIM + d] + rele[r * EMB_DIM + d]
                  - ent[t * EMB_DIM + d];
    e32[idx] = fabsf(v);
}

// ---------------------------------------------------------------------------
// Kernel 2: q[r][j] = e_j^T * W_r * e_j  via  T = E @ W_r^T  (WMMA bf16),
// then per-lane dot with e_j and a 16-lane butterfly reduction.
// One wave per (r, j-tile). 8 v_wmma_f32_16x16x32_bf16 per wave.
// ---------------------------------------------------------------------------
__global__ __launch_bounds__(256) void q_kernel(
    const float* __restrict__ e32,      // (BATCH, 64)
    const float* __restrict__ relw,     // (REL_SIZE, 64, 64) row-major [d][k]
    float* __restrict__ q)              // (REL_SIZE, BATCH) out
{
    const int lane = threadIdx.x & 31;
    const int wave = threadIdx.x >> 5;
    const int task = blockIdx.x * WAVES_PER_BLOCK + wave;   // uniform per wave
    if (task >= NTASK) return;

    const int r    = task >> 9;        // task / JTILES  (r-major -> L2 reuse of W_r)
    const int jt   = task & (JTILES - 1);
    const int half = lane >> 4;        // 0: lanes 0-15, 1: lanes 16-31
    const int ln   = lane & 15;
    const int j0   = jt * 16;

    v8f acc[4] = {v8f{}, v8f{}, v8f{}, v8f{}};

    const float* erow = e32 + (size_t)(j0 + ln) * EMB_DIM;   // A row: E[j0+ln][*]

#pragma unroll
    for (int kc = 0; kc < 2; ++kc) {
        const int k0 = kc * 32;

        // ---- A fragment: 16x32 bf16, E rows j0..j0+15 ----
        // ISA layout: lane M = ln; VGPR v, elem e:
        //   K = k0 + (v<4 ? 0 : 16) + (v&3)*2 + e + 8*half
        v16bf a;
#pragma unroll
        for (int v = 0; v < 8; ++v) {
            const int k = k0 + ((v < 4) ? 0 : 16) + (v & 3) * 2 + half * 8;
            const float2 f = *(const float2*)(erow + k);
            a[2 * v]     = (__bf16)f.x;
            a[2 * v + 1] = (__bf16)f.y;
        }

        // ---- B fragments: 32x16 bf16, B[k][n] = W_r[n][k] (so T = E @ W_r^T) ----
        // ISA layout: lane N = ln; VGPR v, elem e: K = k0 + (v<4 ? 0 : 16) + (v&3)*2 + e + 8*half
#pragma unroll
        for (int nt = 0; nt < 4; ++nt) {
            const int d = nt * 16 + ln;
            const float* wrow = relw + ((size_t)r * EMB_DIM + d) * EMB_DIM;
            v16bf b;
#pragma unroll
            for (int v = 0; v < 8; ++v) {
                const int k = k0 + ((v < 4) ? 0 : 16) + (v & 3) * 2 + half * 8;
                const float2 f = *(const float2*)(wrow + k);
                b[2 * v]     = (__bf16)f.x;
                b[2 * v + 1] = (__bf16)f.y;
            }
            acc[nt] = __builtin_amdgcn_wmma_f32_16x16x32_bf16(
                false, a, false, b, (short)0, acc[nt], false, false);
        }
    }

    // ---- q[j] = sum_d T[j][d] * e[j][d] ----
    // D layout: lane holds T[j0 + rr + 8*half][nt*16 + ln] in acc[nt][rr].
#pragma unroll
    for (int rr = 0; rr < 8; ++rr) {
        const int j = j0 + rr + 8 * half;
        float s = 0.0f;
#pragma unroll
        for (int nt = 0; nt < 4; ++nt) {
            s += acc[nt][rr] * e32[(size_t)j * EMB_DIM + nt * 16 + ln];
        }
        // reduce across the 16 lanes of this half (xor masks stay in-half)
        s += __shfl_xor(s, 1, 32);
        s += __shfl_xor(s, 2, 32);
        s += __shfl_xor(s, 4, 32);
        s += __shfl_xor(s, 8, 32);
        if (ln == 0) q[(size_t)r * BATCH + j] = s;
    }
}

// ---------------------------------------------------------------------------
// Kernel 3: out[i][:] = q[rel_i][:]   -- streaming 268 MB write (the roofline)
// ---------------------------------------------------------------------------
__global__ __launch_bounds__(256) void scatter_kernel(
    const int* __restrict__ trip,
    const float* __restrict__ q,
    float* __restrict__ out)
{
    const int i = blockIdx.x;
    const int r = trip[i * 3 + 1];
    const float4* __restrict__ src = (const float4*)(q + (size_t)r * BATCH);
    float4* __restrict__ dst = (float4*)(out + (size_t)i * BATCH);
#pragma unroll 4
    for (int c = threadIdx.x; c < BATCH / 4; c += 256) {
        dst[c] = src[c];
    }
}

// ---------------------------------------------------------------------------
extern "C" void kernel_launch(void* const* d_in, const int* in_sizes, int n_in,
                              void* d_out, int out_size, void* d_ws, size_t ws_size,
                              hipStream_t stream) {
    const int*   trip = (const int*)d_in[0];     // (8192,3) int32
    const float* ent  = (const float*)d_in[1];   // (1e6,64) f32
    const float* rele = (const float*)d_in[2];   // (1000,64) f32
    const float* relw = (const float*)d_in[3];   // (1000,64,64) f32
    float* out = (float*)d_out;                  // (8192,8192) f32

    // ws layout: e32 (8192*64 f32 = 2 MB) | q (1000*8192 f32 = 32.8 MB)
    float* e32 = (float*)d_ws;
    float* q   = e32 + (size_t)BATCH * EMB_DIM;

    err_kernel<<<(BATCH * EMB_DIM) / 256, 256, 0, stream>>>(trip, ent, rele, e32);

    const int nblocks = (NTASK + WAVES_PER_BLOCK - 1) / WAVES_PER_BLOCK;  // 64,000
    q_kernel<<<nblocks, 256, 0, stream>>>(e32, relw, q);

    scatter_kernel<<<BATCH, 256, 0, stream>>>(trip, q, out);
}
